// VectorQuantizerEMA_678604832942
// MI455X (gfx1250) — matrix operand order repaired and checked by reading, not compiled, not run on hardware
//
#include <hip/hip_runtime.h>

typedef __attribute__((ext_vector_type(16))) _Float16     v16h;
typedef __attribute__((ext_vector_type(8)))  float        v8f;
typedef __attribute__((ext_vector_type(8)))  unsigned int v8u;

static constexpr int K_EMB     = 1024;
static constexpr int D_DIM     = 256;
static constexpr int B_SZ      = 64;
static constexpr int HW_SZ     = 1024;                  // 32*32
static constexpr int CHW_SZ    = D_DIM * HW_SZ;         // 262144
static constexpr int N_ROWS    = B_SZ * HW_SZ;          // 65536
static constexpr int ROWS_PB   = 256;                   // rows per block (32/wave)
static constexpr int GRID_MAIN = N_ROWS / ROWS_PB;      // 256
static constexpr int XP        = 264;                   // padded half pitch (bank spread)

// workspace layout (bytes)
static constexpr size_t WS_ENORM  = 0;                  // float[1024]
static constexpr size_t WS_COUNTS = 4096;               // uint[1024]
static constexpr size_t WS_PART   = 8192;               // float[256]
static constexpr size_t WS_EMB16  = 10240;              // _Float16[1024*256]

// --------------------------------------------------------------------------
__global__ void vq_zero_kernel(unsigned int* __restrict__ counts) {
    counts[blockIdx.x * 256 + threadIdx.x] = 0u;
}

// --------------------------------------------------------------------------
// Per-code: f32 -> f16 codebook copy + ||e||^2
__global__ void vq_prep_kernel(const float* __restrict__ emb,
                               float* __restrict__ enorm,
                               _Float16* __restrict__ emb16) {
    __shared__ float red[256];
    const int k = blockIdx.x;
    const int d = threadIdx.x;
    float v = emb[k * D_DIM + d];
    emb16[k * D_DIM + d] = (_Float16)v;
    red[d] = v * v;
    __syncthreads();
    for (int s = 128; s > 0; s >>= 1) {
        if (d < s) red[d] += red[d + s];
        __syncthreads();
    }
    if (d == 0) enorm[k] = red[0];
}

// --------------------------------------------------------------------------
// Main: 256 rows per block, 8 waves x 32 rows (two 16-row A tiles per wave so
// every LDS B-fragment read feeds two WMMAs). Emb chunks staged with
// global_load_async_to_lds_b128 (ASYNCcnt path). Per column tile, all 8 B
// fragments are preloaded before the 16 back-to-back WMMAs so LDS latency is
// overlapped instead of serialized.
__global__ void __launch_bounds__(256)
vq_argmin_kernel(const float* __restrict__ inp,
                 const float* __restrict__ emb,
                 const float* __restrict__ enorm,
                 const _Float16* __restrict__ emb16,
                 unsigned int* __restrict__ counts,
                 float* __restrict__ partials,
                 float* __restrict__ dout) {
    __shared__ _Float16 sx[ROWS_PB][XP];   // x tile    [row][d]  (135 KB)
    __shared__ _Float16 se[128][XP];       // emb chunk [code][d] ( 66 KB)
    __shared__ float    senorm[128];
    __shared__ int      sidx[ROWS_PB];
    __shared__ float    red[256];

    const int t    = threadIdx.x;
    const int lane = t & 31;
    const int wv   = t >> 5;
    const int n0   = blockIdx.x * ROWS_PB;
    const int b    = n0 >> 10;     // batch image
    const int hw0  = n0 & 1023;    // offset inside 32x32 plane (multiple of 256)
    const float* xbase = inp + b * CHW_SZ + hw0;

    // ---- phase 1: stage x tile (NCHW strided -> [row][d] f16 in LDS) ----
    {
        #pragma unroll 4
        for (int d = 0; d < D_DIM; d += 2) {
            _Float16 h0 = (_Float16)xbase[d * HW_SZ + t];        // coalesced
            _Float16 h1 = (_Float16)xbase[(d + 1) * HW_SZ + t];  // coalesced
            unsigned pk = (unsigned)__builtin_bit_cast(unsigned short, h0) |
                          ((unsigned)__builtin_bit_cast(unsigned short, h1) << 16);
            *(unsigned*)&sx[t][d] = pk;
        }
    }
    __syncthreads();

    // ---- per-wave A fragments: 2 tiles x 16 rows x 256 K (8 chunks of 32) ----
    // A k-order used here (contiguous, B matches): lanes 0-15 row M, VGPR j -> K=2j(,+1);
    // lanes 16-31 same rows with K offset +16.
    v16h afr[2][8];
    {
        const int koff = (lane < 16) ? 0 : 16;
        #pragma unroll
        for (int mt = 0; mt < 2; ++mt) {
            const int r = wv * 32 + mt * 16 + (lane & 15);
            const unsigned int* rowp = (const unsigned int*)(&sx[r][0]);
            #pragma unroll
            for (int kc = 0; kc < 8; ++kc) {
                v8u au{};
                #pragma unroll
                for (int j = 0; j < 8; ++j) {
                    int kk = kc * 32 + koff + 2 * j;
                    au[j] = rowp[kk >> 1];
                }
                afr[mt][kc] = __builtin_bit_cast(v16h, au);
            }
        }
    }

    float bmin[2][8];
    int   bidx[2][8];
    #pragma unroll
    for (int mt = 0; mt < 2; ++mt)
        #pragma unroll
        for (int v = 0; v < 8; ++v) { bmin[mt][v] = 3.4e38f; bidx[mt][v] = 0; }

    const int cN    = lane & 15;               // N index within 16-wide tile
    const int koffB = (lane < 16) ? 0 : 16;    // B k-order matches A: upper lanes hold K+16

    for (int ck = 0; ck < 8; ++ck) {
        if (ck) __syncthreads();               // previous chunk fully consumed
        const int cstart = ck * 128;
        // stage emb f16 chunk: 128 codes x 256 halfs, direct-to-LDS async copies
        {
            const uint4* srcb = (const uint4*)(emb16 + cstart * D_DIM);
            #pragma unroll
            for (int it = 0; it < 16; ++it) {
                int lin = it * 256 + t;
                int c = lin >> 5, u = lin & 31;
                const uint4* gp = srcb + (c * 32 + u);
                unsigned lds_off = (unsigned)(uintptr_t)(&se[c][u * 8]);
                asm volatile("global_load_async_to_lds_b128 %0, %1, off"
                             :: "v"(lds_off), "v"(gp) : "memory");
            }
            if (t < 128) senorm[t] = enorm[cstart + t];
            if (ck + 1 < 8)   // pull next chunk toward L2 (global_prefetch_b8)
                __builtin_prefetch(emb16 + (cstart + 128) * D_DIM + t * 128, 0, 1);
            asm volatile("s_wait_asynccnt 0x0" ::: "memory");
        }
        __syncthreads();

        #pragma unroll 1
        for (int ct = 0; ct < 8; ++ct) {
            const int lc = ct * 16 + cN;       // local code index (this lane's N)
            const unsigned int* crow = (const unsigned int*)(&se[lc][0]);
            // preload ALL 8 B fragments (16 x ds_load_b128) before any WMMA
            v8u bu[8];
            #pragma unroll
            for (int kc = 0; kc < 8; ++kc) {
                #pragma unroll
                for (int j = 0; j < 8; ++j)
                    bu[kc][j] = crow[(kc * 32 + koffB + 2 * j) >> 1];
            }
            // 16 independent WMMAs, back-to-back (no D->A/B hazards)
            v8f acc0 = {}, acc1 = {};
            #pragma unroll
            for (int kc = 0; kc < 8; ++kc) {
                v16h bfr = __builtin_bit_cast(v16h, bu[kc]);
                acc0 = __builtin_amdgcn_wmma_f32_16x16x32_f16(
                    false, afr[0][kc], false, bfr, (short)0, acc0, false, false);
                acc1 = __builtin_amdgcn_wmma_f32_16x16x32_f16(
                    false, afr[1][kc], false, bfr, (short)0, acc1, false, false);
            }
            const float en = senorm[lc];
            const int   cg = cstart + lc;
            // C layout: lane<16 -> M=v, N=lane; lane>=16 -> M=v+8, N=lane-16
            #pragma unroll
            for (int v = 0; v < 8; ++v) {
                float d0 = en - 2.0f * acc0[v];    // ||x||^2 omitted (row const)
                float d1 = en - 2.0f * acc1[v];
                if (d0 < bmin[0][v]) { bmin[0][v] = d0; bidx[0][v] = cg; }
                if (d1 < bmin[1][v]) { bmin[1][v] = d1; bidx[1][v] = cg; }
            }
        }
    }

    // ---- cross-lane argmin reduce within each 16-lane group ----
    #pragma unroll
    for (int mt = 0; mt < 2; ++mt)
        #pragma unroll
        for (int v = 0; v < 8; ++v)
            #pragma unroll
            for (int off = 8; off >= 1; off >>= 1) {
                float om = __shfl_xor(bmin[mt][v], off, 32);
                int   oi = __shfl_xor(bidx[mt][v], off, 32);
                if (om < bmin[mt][v] ||
                    (om == bmin[mt][v] && oi < bidx[mt][v])) {
                    bmin[mt][v] = om; bidx[mt][v] = oi;
                }
            }

    float* oidx = dout + 1 + (size_t)B_SZ * CHW_SZ + 1;  // idx region
    if ((lane & 15) == 0) {
        #pragma unroll
        for (int mt = 0; mt < 2; ++mt) {
            const int rb = wv * 32 + mt * 16 + ((lane == 0) ? 0 : 8);
            #pragma unroll
            for (int v = 0; v < 8; ++v) {
                int rl = rb + v;
                int gi = bidx[mt][v];
                sidx[rl] = gi;
                atomicAdd(&counts[gi], 1u);      // deterministic integer atomics
                oidx[n0 + rl] = (float)gi;
            }
        }
    }
    __syncthreads();

    // ---- phase 3: gather exact f32 code rows, write NCHW, loss partials ----
    float lsum = 0.0f;
    {
        float* qbase = dout + 1 + b * CHW_SZ + hw0;
        const int gi = sidx[t];
        const float* qrow = emb + gi * D_DIM;    // each thread streams one row
        #pragma unroll 4
        for (int d = 0; d < D_DIM; ++d) {
            float q  = qrow[d];                  // L2-resident gather
            float xv = xbase[d * HW_SZ + t];     // coalesced re-read (exact f32)
            float df = q - xv;
            lsum += df * df;
            qbase[d * HW_SZ + t] = q;            // straight-through value == q
        }
    }
    red[t] = lsum;
    __syncthreads();
    for (int s = 128; s > 0; s >>= 1) {
        if (t < s) red[t] += red[t + s];
        __syncthreads();
    }
    if (t == 0) partials[blockIdx.x] = red[0];   // deterministic per-block sum
}

// --------------------------------------------------------------------------
__global__ void vq_finalize_kernel(const unsigned int* __restrict__ counts,
                                   const float* __restrict__ partials,
                                   float* __restrict__ dout) {
    __shared__ float red[256];
    const int t = threadIdx.x;
    red[t] = partials[t];                        // 256 block partials
    __syncthreads();
    for (int s = 128; s > 0; s >>= 1) {
        if (t < s) red[t] += red[t + s];
        __syncthreads();
    }
    float loss = 0.25f * red[0] / 16777216.0f;   // mean over B*H*W*D
    __syncthreads();
    float h = 0.0f;
    for (int k = t; k < K_EMB; k += 256) {
        float p = (float)counts[k] * (1.0f / (float)N_ROWS);
        h += p * logf(p + 1e-10f);
    }
    red[t] = h;
    __syncthreads();
    for (int s = 128; s > 0; s >>= 1) {
        if (t < s) red[t] += red[t + s];
        __syncthreads();
    }
    if (t == 0) {
        dout[0] = loss;
        dout[1 + (size_t)B_SZ * CHW_SZ] = expf(-red[0]);   // perplexity
    }
}

// --------------------------------------------------------------------------
extern "C" void kernel_launch(void* const* d_in, const int* in_sizes, int n_in,
                              void* d_out, int out_size, void* d_ws, size_t ws_size,
                              hipStream_t stream) {
    const float* inp = (const float*)d_in[0];   // [64,256,32,32] f32
    const float* emb = (const float*)d_in[1];   // [1024,256] f32
    float* dout = (float*)d_out;                // [loss | q NCHW | perp | idx]

    char* ws = (char*)d_ws;
    float*        enorm    = (float*)(ws + WS_ENORM);
    unsigned int* counts   = (unsigned int*)(ws + WS_COUNTS);
    float*        partials = (float*)(ws + WS_PART);
    _Float16*     emb16    = (_Float16*)(ws + WS_EMB16);

    hipLaunchKernelGGL(vq_zero_kernel, dim3(4), dim3(256), 0, stream, counts);
    hipLaunchKernelGGL(vq_prep_kernel, dim3(K_EMB), dim3(256), 0, stream,
                       emb, enorm, emb16);
    hipLaunchKernelGGL(vq_argmin_kernel, dim3(GRID_MAIN), dim3(256), 0, stream,
                       inp, emb, enorm, emb16, counts, partials, dout);
    hipLaunchKernelGGL(vq_finalize_kernel, dim3(1), dim3(256), 0, stream,
                       counts, partials, dout);
}
